// LIF_complex_9251359555593
// MI455X (gfx1250) — matrix operand order repaired and checked by reading, not compiled, not run on hardware
//
#include <hip/hip_runtime.h>
#include <hip/hip_fp16.h>

#define T_STEPS     8192
#define NN          2048
#define NWG         128          // workgroups; 16 output rows each
#define ROWS_PER_WG 16
#define THREADS     256          // 8 wave32
#define WAVES       8
#define K_PER_WAVE  (NN / WAVES) // 256
#define WMMA_K      32
#define KITERS      (K_PER_WAVE / WMMA_K) // 8

typedef __attribute__((ext_vector_type(16))) _Float16 v16h;
typedef __attribute__((ext_vector_type(8)))  _Float16 v8h;
typedef __attribute__((ext_vector_type(8)))  float    v8f;

// ---------------------------------------------------------------------------
// Init: convert w -> f16 (L2 resident, 8 MB), seed g buffer 0, zero barrier.
// Runs every kernel_launch call (deterministic; d_ws is not re-poisoned).
// ---------------------------------------------------------------------------
__global__ void lif_init(const float* __restrict__ w, _Float16* __restrict__ w16,
                         const float* __restrict__ g0, _Float16* __restrict__ gbuf,
                         unsigned* __restrict__ bar) {
    size_t idx    = (size_t)blockIdx.x * blockDim.x + threadIdx.x;
    size_t stride = (size_t)gridDim.x * blockDim.x;
    for (size_t i = idx; i < (size_t)NN * NN; i += stride)
        w16[i] = (_Float16)w[i];
    for (size_t i = idx; i < NN; i += stride)
        gbuf[i] = (_Float16)g0[i];          // buffer 0; buffer 1 written before read
    if (idx < 1) bar[0] = 0u;               // monotonic arrival counter
}

// ---------------------------------------------------------------------------
// Persistent LIF kernel: one WG per 16-row tile of w (tile lives in VGPRs),
// one device-wide monotonic-counter barrier per timestep.
// ---------------------------------------------------------------------------
__global__ void __launch_bounds__(THREADS, 1) lif_run(
    const float* __restrict__ x_in, const _Float16* __restrict__ w16,
    const float* __restrict__ v_rest, const float* __restrict__ tau_m,
    const float* __restrict__ tau_g, const float* __restrict__ pre_c_p,
    const float* __restrict__ post_c_p, const float* __restrict__ v0,
    const float* __restrict__ g0, float* __restrict__ out,
    _Float16* __restrict__ gbuf,      // double buffer: 2 * NN halves
    unsigned* __restrict__ bar)       // [0] = monotonic arrival counter
{
    __shared__ float parts[WAVES][ROWS_PER_WG];

    const int tid      = threadIdx.x;
    const int wave     = tid >> 5;
    const int lane     = tid & 31;
    const int half_sel = lane >> 4;      // which K-half this lane holds (A layout)
    const int mrow     = lane & 15;      // matrix row within 16-row tile
    const int wg       = blockIdx.x;

    // ---- preload this wave's 16x256 w tile into 64 VGPRs (A operands) -----
    // 16-bit A 16x32 layout: lanes 0-15 hold K {0..7}U{16..23},
    //                        lanes 16-31 hold K {8..15}U{24..31}
    const _Float16* wrow = w16 + (size_t)(wg * ROWS_PER_WG + mrow) * NN
                               + wave * K_PER_WAVE + half_sel * 8;
    v16h aReg[KITERS];
#pragma unroll
    for (int i = 0; i < KITERS; ++i) {
        v8h alo = *(const v8h*)(wrow + i * WMMA_K);
        v8h ahi = *(const v8h*)(wrow + i * WMMA_K + 16);
        aReg[i] = __builtin_shufflevector(alo, ahi,
                  0,1,2,3,4,5,6,7,8,9,10,11,12,13,14,15);
    }

    // ---- per-neuron persistent state in registers of threads 0..15 --------
    float v_s = 0.f, g_s = 0.f, vr = 0.f, inv_tm = 0.f, inv_tg = 0.f;
    float pre_c = 0.f, post_c = 0.f, xt = 0.f;
    const int n = wg * ROWS_PER_WG + tid;    // valid when tid < 16
    if (tid < ROWS_PER_WG) {
        v_s    = v0[n];
        g_s    = g0[n];
        vr     = v_rest[n];
        inv_tm = 1.0f / tau_m[n];
        inv_tg = 1.0f / tau_g[n];
        pre_c  = pre_c_p[0];
        post_c = post_c_p[0];
        xt     = x_in[n];                    // x for t = 0
    }

    for (int t = 0; t < T_STEPS; ++t) {
        // acquire: pairs with the relaxed spin load at the end of the previous
        // step; invalidates this wave's CU caches so fresh g is read from L2
        __builtin_amdgcn_fence(__ATOMIC_ACQUIRE, "agent");

        // prefetch next step's x; consumed only next iteration, so its
        // latency hides under this whole step including the barrier
        float xt_next = 0.f;
        if (tid < ROWS_PER_WG && t + 1 < T_STEPS)
            xt_next = x_in[(size_t)(t + 1) * NN + n];

        // ---- issue ALL B loads (L2-hot, broadcast within half-wave) ------
        // B 32x16 layout: lanes 0-15 hold K {0..7}U{16..23}, lanes 16-31 hold K {8..15}U{24..31}
        const _Float16* bbase = gbuf + (size_t)(t & 1) * NN
                                     + wave * K_PER_WAVE + half_sel * 8;
        v16h bReg[KITERS];
#pragma unroll
        for (int i = 0; i < KITERS; ++i) {
            v8h blo = *(const v8h*)(bbase + i * WMMA_K);
            v8h bhi = *(const v8h*)(bbase + i * WMMA_K + 16);
            bReg[i] = __builtin_shufflevector(blo, bhi,
                      0,1,2,3,4,5,6,7,8,9,10,11,12,13,14,15);
        }

        // Data-dependence scheduling barrier: ties all 8 B vectors through a
        // no-op asm. Loads must issue above it (they define the operands),
        // WMMAs must stay below it (they consume the asm results), and all
        // 64 B VGPRs stay live -> one s_wait_loadcnt for the whole batch,
        // i.e. exactly one L2 round-trip per timestep.
        asm volatile(""
            : "+v"(bReg[0]), "+v"(bReg[1]), "+v"(bReg[2]), "+v"(bReg[3]),
              "+v"(bReg[4]), "+v"(bReg[5]), "+v"(bReg[6]), "+v"(bReg[7]));

        // ---- 8 back-to-back WMMAs, f32 accumulation ----------------------
        v8f acc = {};
#pragma unroll
        for (int i = 0; i < KITERS; ++i)
            acc = __builtin_amdgcn_wmma_f32_16x16x32_f16(
                      false, aReg[i], false, bReg[i], (short)0, acc, false, false);

        // D layout: lane 0 holds rows 0-7 (col 0), lane 16 holds rows 8-15
        if (mrow == 0) {
#pragma unroll
            for (int r = 0; r < 8; ++r)
                parts[wave][half_sel * 8 + r] = acc[r];
        }
        __syncthreads();

        // ---- neuron update (threads 0..15 own one neuron each) -----------
        if (tid < ROWS_PER_WG) {
            float y = 0.f;
#pragma unroll
            for (int wv = 0; wv < WAVES; ++wv) y += parts[wv][tid];

            const float I = post_c / (1.0f + __expf(-pre_c * (y + xt)));
            g_s = g_s - g_s * inv_tg;
            v_s = v_s + (vr - v_s + I) * inv_tm;
            const float soft = 1.0f / (1.0f + __expf(-(v_s - 30.0f)));
            const float spk  = (v_s >= 30.0f) ? 1.0f : 0.0f;
            v_s = spk * vr + (1.0f - spk) * v_s;
            g_s = spk + (1.0f - spk) * g_s;

            out[(size_t)t * NN + n]                        = v_s;   // vs (post-reset)
            out[(size_t)T_STEPS * NN + (size_t)t * NN + n] = soft;  // soft spikes
            gbuf[(size_t)((t + 1) & 1) * NN + n] = (_Float16)g_s;   // publish next g
            __builtin_amdgcn_fence(__ATOMIC_RELEASE, "agent");      // flush g to L2
        }
        xt = xt_next;
        __syncthreads();

        // ---- device-wide barrier: single monotonic counter ---------------
        if (tid == 0) {
            __hip_atomic_fetch_add(&bar[0], 1u, __ATOMIC_RELEASE,
                                   __HIP_MEMORY_SCOPE_AGENT);
            const unsigned target = (unsigned)(t + 1) * NWG;
            // relaxed poll; the fence(acquire) at the top of the next step
            // provides the required load->fence-acquire ordering
            while (__hip_atomic_load(&bar[0], __ATOMIC_RELAXED,
                                     __HIP_MEMORY_SCOPE_AGENT) < target) {
                __builtin_amdgcn_s_sleep(2);
            }
        }
        __syncthreads();
    }
}

// ---------------------------------------------------------------------------
extern "C" void kernel_launch(void* const* d_in, const int* in_sizes, int n_in,
                              void* d_out, int out_size, void* d_ws, size_t ws_size,
                              hipStream_t stream) {
    const float* x_in   = (const float*)d_in[0];
    const float* w      = (const float*)d_in[1];
    const float* v_rest = (const float*)d_in[2];
    const float* tau_m  = (const float*)d_in[3];
    const float* tau_g  = (const float*)d_in[4];
    const float* pre_c  = (const float*)d_in[5];
    const float* post_c = (const float*)d_in[6];
    const float* v0     = (const float*)d_in[7];
    const float* g0     = (const float*)d_in[8];
    float* out = (float*)d_out;

    char* ws = (char*)d_ws;
    _Float16* w16  = (_Float16*)ws;                                  // 8 MB
    _Float16* gbuf = (_Float16*)(ws + (size_t)NN * NN * 2);          // 8 KB
    unsigned* bar  = (unsigned*)(ws + (size_t)NN * NN * 2 + 2 * NN * 2);

    lif_init<<<256, 256, 0, stream>>>(w, w16, g0, gbuf, bar);
    lif_run<<<NWG, THREADS, 0, stream>>>(x_in, w16, v_rest, tau_m, tau_g,
                                         pre_c, post_c, v0, g0, out, gbuf, bar);
}
